// MultiLoRALinear_15187004358681
// MI455X (gfx1250) — matrix operand order repaired and checked by reading, not compiled, not run on hardware
//
#include <hip/hip_runtime.h>
#include <stdint.h>

#define CDNA5_USE_TDM 1

// ---------------- problem dims (fixed by the reference) ----------------
constexpr int B_ = 8;
constexpr int T_ = 2048;
constexpr int M_ = B_ * T_;   // 16384 rows
constexpr int K_ = 4096;
constexpr int N_ = 4096;
constexpr int NL_ = 16;       // experts
constexpr int R_ = 32;        // rank
constexpr int RC_ = 64;       // combined rank: 32 expert + 32 shared

// ---------------- vector types for WMMA ----------------
typedef __attribute__((ext_vector_type(16))) __bf16 v16bf;
typedef __attribute__((ext_vector_type(8)))  float  v8f;
typedef __attribute__((ext_vector_type(4)))  unsigned int v4u;
typedef __attribute__((ext_vector_type(8)))  int    v8i;
typedef __attribute__((ext_vector_type(4)))  int    v4i;

// ---------------- bf16 helpers (round-to-nearest-even) ----------------
__device__ __forceinline__ unsigned short f2bf(float f) {
  unsigned u = __builtin_bit_cast(unsigned, f);
  return (unsigned short)((u + 0x7FFFu + ((u >> 16) & 1u)) >> 16);
}
__device__ __forceinline__ float bf2f(unsigned short s) {
  unsigned u = ((unsigned)s) << 16;
  return __builtin_bit_cast(float, u);
}

// Build a v16bf fragment from two 16-byte chunks (works for LDS or global ptrs).
struct uint4x2 { uint4 a, b; };
__device__ __forceinline__ v16bf frag_pair(const void* p0, const void* p1) {
  uint4x2 t;
  t.a = *(const uint4*)p0;
  t.b = *(const uint4*)p1;
  return __builtin_bit_cast(v16bf, t);
}

__device__ __forceinline__ v8f wmma_bf16(v16bf a, v16bf b, v8f c) {
  // D = A*B + C, 16x16x32 bf16 -> f32
  return __builtin_amdgcn_wmma_f32_16x16x32_bf16(false, a, false, b, (short)0, c,
                                                 false, false);
}

// ---------------- GEMM tiling ----------------
constexpr int BM = 128, BN = 128, KS = 64;
constexpr int PITCH = 144;              // LDS row pitch bytes: 128B data + 16B pad (36 banks)
constexpr int TILEB = BM * PITCH;       // 18432 B per staged tile
constexpr int NST = K_ / KS;            // 64 K-stages

#if CDNA5_USE_TDM
// TDM 2D tile load: 128 rows x 64 bf16, tensor row stride 4096 elems,
// LDS padding 16B per 128B row (pad_interval=32 dwords -> code 4, pad=4 dwords -> code 3).
__device__ __forceinline__ void tdm_load_tile(unsigned lds_off, const void* gptr) {
  unsigned long long ga = (unsigned long long)(uintptr_t)gptr;
  v4u g0;
  g0[0] = 1u;                                        // count = 1 valid descriptor
  g0[1] = lds_off;                                   // LDS byte address
  g0[2] = (unsigned)ga;                              // global addr lo
  g0[3] = (unsigned)((ga >> 32) & 0x01FFFFFFull) | (2u << 30);  // addr hi | type=2
  v8i g1;
  g1[0] = (1 << 16) | (1 << 20) | (4 << 22) | (3 << 25); // data_size=2B, pad_en, interval, amount
  g1[1] = (int)((4096u & 0xFFFFu) << 16);            // tensor_dim0[15:0] @ bits63:48
  g1[2] = (int)((4096u >> 16) | ((128u & 0xFFFFu) << 16)); // dim0 hi | tensor_dim1 lo
  g1[3] = (int)((0u) | (64u << 16));                 // tensor_dim1 hi | tile_dim0=64
  g1[4] = (int)(128u);                               // tile_dim1=128, tile_dim2=0
  g1[5] = (int)4096;                                 // tensor_dim0_stride lo32 (elems)
  g1[6] = 0;
  g1[7] = 0;
  v4i z4; z4[0] = 0; z4[1] = 0; z4[2] = 0; z4[3] = 0;
  v8i z8;
  #pragma unroll
  for (int i = 0; i < 8; ++i) z8[i] = 0;
  // 6-arg form (clang-23 / therock-10.0 lane): groups 0..3 (+extra), cpol
  __builtin_amdgcn_tensor_load_to_lds(g0, g1, z4, z4, z8, 0);
}
#else
__device__ __forceinline__ void stage_tile(char* dst, const unsigned short* __restrict__ src,
                                           int tid) {
  #pragma unroll
  for (int i = 0; i < 4; ++i) {
    int c = tid + i * 256;           // 1024 chunks of 8 elems
    int row = c >> 3, col = (c & 7) << 3;
    *(uint4*)(dst + (size_t)row * PITCH + col * 2) =
        *(const uint4*)(src + (size_t)row * K_ + col);
  }
}
#endif

// ---------------- pre-pass kernels ----------------
__global__ void pack_x_kernel(const float* __restrict__ x, unsigned short* __restrict__ hi,
                              unsigned short* __restrict__ lo, int do_lo, int n4) {
  int i = blockIdx.x * 256 + threadIdx.x;
  if (i >= n4) return;
  float4 v = ((const float4*)x)[i];
  ushort4 h;
  h.x = f2bf(v.x); h.y = f2bf(v.y); h.z = f2bf(v.z); h.w = f2bf(v.w);
  ((ushort4*)hi)[i] = h;
  if (do_lo) {
    ushort4 l;
    l.x = f2bf(v.x - bf2f(h.x)); l.y = f2bf(v.y - bf2f(h.y));
    l.z = f2bf(v.z - bf2f(h.z)); l.w = f2bf(v.w - bf2f(h.w));
    ((ushort4*)lo)[i] = l;
  }
}

// W (K x N row-major) -> Wt (N x K) bf16 hi/lo, 64x64 LDS transpose tiles
__global__ void pack_wt_kernel(const float* __restrict__ W, unsigned short* __restrict__ hi,
                               unsigned short* __restrict__ lo, int do_lo) {
  __shared__ float t[64][65];
  const int n0 = blockIdx.x * 64, k0 = blockIdx.y * 64;
  const int tx = threadIdx.x, ty = threadIdx.y;   // (64,4)
  for (int r = ty; r < 64; r += 4)
    t[r][tx] = W[(size_t)(k0 + r) * N_ + n0 + tx];
  __syncthreads();
  for (int r = ty; r < 64; r += 4) {
    float v = t[tx][r];
    unsigned short h = f2bf(v);
    size_t o = (size_t)(n0 + r) * K_ + k0 + tx;
    hi[o] = h;
    if (do_lo) lo[o] = f2bf(v - bf2f(h));
  }
}

// A_comb[b][r][k]: r<32 -> sum_e gate[b,e]*lora_A[e,r,k]; r>=32 -> A_shared[r-32,k]  (fp32)
__global__ void build_Ac_kernel(const float* __restrict__ gate, const float* __restrict__ lA,
                                const float* __restrict__ lAs, float* __restrict__ Ac) {
  int idx = blockIdx.x * 256 + threadIdx.x;     // 8*64*4096 = 2M
  int k = idx & (K_ - 1);
  int rr = idx >> 12;
  int r = rr & 63, b = rr >> 6;
  float v;
  if (r < R_) {
    v = 0.f;
    #pragma unroll
    for (int e = 0; e < NL_; ++e)
      v += gate[b * NL_ + e] * lA[((size_t)(e * R_ + r)) * K_ + k];
  } else {
    v = lAs[(size_t)(r - R_) * K_ + k];
  }
  Ac[idx] = v;
}

// Bc[b][n][r] bf16: r<32 -> sum_e gate*lora_B[e,n,r]; r>=32 -> B_shared[n,r-32]
__global__ void build_Bc_kernel(const float* __restrict__ gate, const float* __restrict__ lB,
                                const float* __restrict__ lBs, unsigned short* __restrict__ Bc) {
  int idx = blockIdx.x * 256 + threadIdx.x;     // 8*4096*64 = 2M
  int r = idx & 63;
  int n = (idx >> 6) & (N_ - 1);
  int b = idx >> 18;
  float v;
  if (r < R_) {
    v = 0.f;
    #pragma unroll
    for (int e = 0; e < NL_; ++e)
      v += gate[b * NL_ + e] * lB[((size_t)(e * N_ + n)) * R_ + r];
  } else {
    v = lBs[(size_t)n * R_ + (r - R_)];
  }
  Bc[idx] = f2bf(v);
}

// Ax[m][r] = sum_k x[m,k] * Ac[b][r][k]   (fp32 accumulate, bf16 out). One row per block.
__global__ void compute_Ax_kernel(const float* __restrict__ x, const float* __restrict__ Ac,
                                  unsigned short* __restrict__ Axb) {
  __shared__ float xr[K_];
  __shared__ float red[256];
  const int m = blockIdx.x;
  const int b = m >> 11;                 // T = 2048
  const int tid = threadIdx.x;
  for (int k = tid; k < K_; k += 256) xr[k] = x[(size_t)m * K_ + k];
  __syncthreads();
  const int r = tid & 63, c = tid >> 6;  // 4 K-chunks of 1024
  const float* a = Ac + ((size_t)b * RC_ + r) * K_ + c * 1024;
  const float* xs = xr + c * 1024;
  float s = 0.f;
  for (int kk = 0; kk < 1024; ++kk) s += xs[kk] * a[kk];
  red[tid] = s;
  __syncthreads();
  if (c == 0)
    Axb[(size_t)m * RC_ + r] = f2bf(red[r] + red[64 + r] + red[128 + r] + red[192 + r]);
}

// ---------------- fused GEMM + LoRA epilogue ----------------
template <bool SPLIT>
__global__ __launch_bounds__(256)
void gemm_lora_kernel(const unsigned short* __restrict__ Xh, const unsigned short* __restrict__ Xl,
                      const unsigned short* __restrict__ Wh, const unsigned short* __restrict__ Wl,
                      const unsigned short* __restrict__ Axb, const unsigned short* __restrict__ Bcb,
                      const float* __restrict__ bias, float* __restrict__ out) {
  extern __shared__ char lds[];
  constexpr int NTILE = SPLIT ? 4 : 2;
  constexpr int SS = NTILE * TILEB;

  const int tid = threadIdx.x;
  const int wid = tid >> 5;
  const int lane = tid & 31;
  const int lrow = lane & 15;
  const int lhi = lane >> 4;
  const int wm = (wid >> 2) * 64;   // wave M offset within block tile
  const int wn = (wid & 3) * 32;    // wave N offset
  const int m0 = blockIdx.y * BM;
  const int n0 = blockIdx.x * BN;
  const int bb = m0 >> 11;          // batch (BM divides T)

  v8f acc[4][2];
  #pragma unroll
  for (int i = 0; i < 4; ++i)
    #pragma unroll
    for (int j = 0; j < 2; ++j) acc[i][j] = (v8f)(0.0f);

#if CDNA5_USE_TDM
  const unsigned ldsbase = (unsigned)(unsigned long long)(uintptr_t)lds;
  auto issue = [&](int s) {
    const size_t k0 = (size_t)s * KS;
    const unsigned base = ldsbase + (unsigned)((s & 1) * SS);
    tdm_load_tile(base,             (const void*)(Xh + (size_t)m0 * K_ + k0));
    tdm_load_tile(base + TILEB,     (const void*)(Wh + (size_t)n0 * K_ + k0));
    if (SPLIT) {
      tdm_load_tile(base + 2 * TILEB, (const void*)(Xl + (size_t)m0 * K_ + k0));
      tdm_load_tile(base + 3 * TILEB, (const void*)(Wl + (size_t)n0 * K_ + k0));
    }
  };
  if (__builtin_amdgcn_readfirstlane((int)tid) < 32) issue(0);
#endif

  for (int s = 0; s < NST; ++s) {
#if CDNA5_USE_TDM
    if (__builtin_amdgcn_readfirstlane((int)tid) < 32) {
      if (s + 1 < NST) {
        issue(s + 1);                                  // prefetch next stage
        __builtin_amdgcn_s_wait_tensorcnt(NTILE);      // current stage complete
      } else {
        __builtin_amdgcn_s_wait_tensorcnt(0);
      }
    }
    __syncthreads();
    const char* Ab = lds + (s & 1) * SS;
#else
    {
      const size_t k0 = (size_t)s * KS;
      stage_tile(lds,          Xh + (size_t)m0 * K_ + k0, tid);
      stage_tile(lds + TILEB,  Wh + (size_t)n0 * K_ + k0, tid);
      if (SPLIT) {
        stage_tile(lds + 2 * TILEB, Xl + (size_t)m0 * K_ + k0, tid);
        stage_tile(lds + 3 * TILEB, Wl + (size_t)n0 * K_ + k0, tid);
      }
    }
    __syncthreads();
    const char* Ab = lds;
#endif
    const char* Bb = Ab + TILEB;

    #pragma unroll
    for (int kc = 0; kc < 2; ++kc) {       // two 16x16x32 chunks per 64-wide stage
      const int ko = kc * 32;
      v16bf af[4], bfv[2];
      #pragma unroll
      for (int mi = 0; mi < 4; ++mi) {
        const char* p = Ab + (size_t)(wm + mi * 16 + lrow) * PITCH + (ko + lhi * 8) * 2;
        af[mi] = frag_pair(p, p + 32);     // K 0-7 & 16-23 (lanes<16) / 8-15 & 24-31
      }
      #pragma unroll
      for (int ni = 0; ni < 2; ++ni) {
        const char* p = Bb + (size_t)(wn + ni * 16 + lrow) * PITCH + (ko + lhi * 8) * 2;
        bfv[ni] = frag_pair(p, p + 32);    // K 0-7 & 16-23 (lanes<16) / 8-15 & 24-31
      }
      #pragma unroll
      for (int mi = 0; mi < 4; ++mi)
        #pragma unroll
        for (int ni = 0; ni < 2; ++ni)
          acc[mi][ni] = wmma_bf16(af[mi], bfv[ni], acc[mi][ni]);

      if constexpr (SPLIT) {               // hi*lo + lo*hi correction terms
        const char* Al = Ab + 2 * TILEB;
        const char* Bl = Ab + 3 * TILEB;
        v16bf afl[4], bfl[2];
        #pragma unroll
        for (int mi = 0; mi < 4; ++mi) {
          const char* p = Al + (size_t)(wm + mi * 16 + lrow) * PITCH + (ko + lhi * 8) * 2;
          afl[mi] = frag_pair(p, p + 32);
        }
        #pragma unroll
        for (int ni = 0; ni < 2; ++ni) {
          const char* p = Bl + (size_t)(wn + ni * 16 + lrow) * PITCH + (ko + lhi * 8) * 2;
          bfl[ni] = frag_pair(p, p + 32);
        }
        #pragma unroll
        for (int mi = 0; mi < 4; ++mi)
          #pragma unroll
          for (int ni = 0; ni < 2; ++ni) {
            acc[mi][ni] = wmma_bf16(af[mi], bfl[ni], acc[mi][ni]);
            acc[mi][ni] = wmma_bf16(afl[mi], bfv[ni], acc[mi][ni]);
          }
      }
    }
    __syncthreads();
  }

  // ---- LoRA epilogue: two extra rank-32 WMMA chunks straight from global (L2-resident) ----
  {
    const unsigned short* Ax0 = Axb + (size_t)m0 * RC_;
    const unsigned short* Bc0 = Bcb + ((size_t)bb * N_ + n0) * RC_;
    #pragma unroll
    for (int rc = 0; rc < 2; ++rc) {
      const int ko = rc * 32;
      v16bf af[4], bfv[2];
      #pragma unroll
      for (int mi = 0; mi < 4; ++mi) {
        const unsigned short* p = Ax0 + (size_t)(wm + mi * 16 + lrow) * RC_ + ko + lhi * 8;
        af[mi] = frag_pair(p, p + 16);
      }
      #pragma unroll
      for (int ni = 0; ni < 2; ++ni) {
        const unsigned short* p = Bc0 + (size_t)(wn + ni * 16 + lrow) * RC_ + ko + lhi * 8;
        bfv[ni] = frag_pair(p, p + 16);
      }
      #pragma unroll
      for (int mi = 0; mi < 4; ++mi)
        #pragma unroll
        for (int ni = 0; ni < 2; ++ni)
          acc[mi][ni] = wmma_bf16(af[mi], bfv[ni], acc[mi][ni]);
    }
  }

  // ---- store: C layout M = vgpr + 8*(lane>=16), N = lane&15 ----
  #pragma unroll
  for (int mi = 0; mi < 4; ++mi)
    #pragma unroll
    for (int ni = 0; ni < 2; ++ni) {
      const int nn = n0 + wn + ni * 16 + lrow;
      const float bv = bias[nn];
      const int mb = m0 + wm + mi * 16 + lhi * 8;
      float* op = out + (size_t)mb * N_ + nn;
      v8f c = acc[mi][ni];
      #pragma unroll
      for (int v = 0; v < 8; ++v) op[(size_t)v * N_] = c[v] + bv;
    }
}

// ---------------- host launcher ----------------
extern "C" void kernel_launch(void* const* d_in, const int* in_sizes, int n_in,
                              void* d_out, int out_size, void* d_ws, size_t ws_size,
                              hipStream_t stream) {
  (void)in_sizes; (void)n_in; (void)out_size;
  const float* x    = (const float*)d_in[0];
  const float* gate = (const float*)d_in[1];
  const float* W    = (const float*)d_in[2];
  const float* bias = (const float*)d_in[3];
  const float* lA   = (const float*)d_in[4];
  const float* lB   = (const float*)d_in[5];
  const float* lAs  = (const float*)d_in[6];
  const float* lBs  = (const float*)d_in[7];
  float* out = (float*)d_out;

  char* ws = (char*)d_ws;
  size_t o = 0;
  auto carve = [&](size_t bytes) {
    char* p = ws + o;
    o += (bytes + 255) & ~(size_t)255;
    return p;
  };
  unsigned short* xh  = (unsigned short*)carve((size_t)M_ * K_ * 2);   // 128 MiB
  unsigned short* wh  = (unsigned short*)carve((size_t)K_ * N_ * 2);   //  32 MiB
  unsigned short* axb = (unsigned short*)carve((size_t)M_ * RC_ * 2);
  unsigned short* bcb = (unsigned short*)carve((size_t)B_ * N_ * RC_ * 2);
  float*          ac  = (float*)carve((size_t)B_ * RC_ * K_ * 4);
  unsigned short* xl  = (unsigned short*)carve((size_t)M_ * K_ * 2);
  unsigned short* wl  = (unsigned short*)carve((size_t)K_ * N_ * 2);
  const bool split = ws_size >= o;           // bf16x3 split path if lo-planes fit

  const int n4 = (M_ * K_) / 4;
  pack_x_kernel<<<n4 / 256, 256, 0, stream>>>(x, xh, xl, split ? 1 : 0, n4);
  pack_wt_kernel<<<dim3(N_ / 64, K_ / 64), dim3(64, 4), 0, stream>>>(W, wh, wl, split ? 1 : 0);
  build_Ac_kernel<<<(B_ * RC_ * K_) / 256, 256, 0, stream>>>(gate, lA, lAs, ac);
  build_Bc_kernel<<<(B_ * N_ * RC_) / 256, 256, 0, stream>>>(gate, lB, lBs, bcb);
  compute_Ax_kernel<<<M_, 256, 0, stream>>>(x, ac, axb);

  if (split) {
    const size_t shmem = (size_t)4 * TILEB * 2;   // 144 KiB double-buffered
    (void)hipFuncSetAttribute((const void*)gemm_lora_kernel<true>,
                              hipFuncAttributeMaxDynamicSharedMemorySize, (int)shmem);
    gemm_lora_kernel<true><<<dim3(N_ / BN, M_ / BM), 256, shmem, stream>>>(
        xh, xl, wh, wl, axb, bcb, bias, out);
  } else {
    const size_t shmem = (size_t)2 * TILEB * 2;   // 72 KiB double-buffered
    (void)hipFuncSetAttribute((const void*)gemm_lora_kernel<false>,
                              hipFuncAttributeMaxDynamicSharedMemorySize, (int)shmem);
    gemm_lora_kernel<false><<<dim3(N_ / BN, M_ / BM), 256, shmem, stream>>>(
        xh, xl, wh, wl, axb, bcb, bias, out);
  }
}
